// SampleModel_60309930770455
// MI455X (gfx1250) — matrix operand order repaired and checked by reading, not compiled, not run on hardware
//
#include <hip/hip_runtime.h>
#include <hip/hip_bf16.h>
#include <cmath>

typedef __attribute__((ext_vector_type(16))) _Float16 v16h;
typedef __attribute__((ext_vector_type(8)))  float    v8f;

// A-fragment K index for 16-bit A 16x32 layout: element i, lane-half h
__device__ __forceinline__ int a_k(int h, int i) {
    return (i < 8) ? (h * 8 + i) : (16 + h * 8 + (i - 8));
}
__device__ __forceinline__ int imin(int a, int b) { return a < b ? a : b; }

// Async memory->LDS copy of 16 bytes (GLOBAL_LOAD_ASYNC_TO_LDS_B128,
// tracked by ASYNCcnt). lds must be 16B aligned; gptr 16B aligned.
__device__ __forceinline__ void async_ld_b128(void* lds, const void* gptr) {
    unsigned loff = (unsigned)(uintptr_t)lds;   // low 32 bits = LDS byte addr
    asm volatile("global_load_async_to_lds_b128 %0, %1, off"
                 :: "v"(loff), "v"(gptr) : "memory");
}
__device__ __forceinline__ void wait_asynccnt0() {
    asm volatile("s_wait_asynccnt 0x0" ::: "memory");
}

// ---------------------------------------------------------------------------
// K1: conv1 (7x7, 1->8) as implicit GEMM via WMMA, then 2x2 maxpool+bias+relu
//     grid = B blocks, 256 threads (8 waves). M=484 conv positions (31 tiles),
//     K=49 pad 64 (2 chunks of 32), N=8 pad 16.
//     x tile staged with async LDS loads; im2col gathers are unconditional
//     (xs zero-padded so every address for r<496, k<64 is in-bounds; B is
//     zero for k>=49).
// ---------------------------------------------------------------------------
__global__ __launch_bounds__(256) void k1_conv1(const float* __restrict__ x,
                                                const float* __restrict__ w,
                                                const float* __restrict__ bias,
                                                float* __restrict__ h1) {
    __shared__ __align__(16) float xs[904];  // 28x28 input + zero pad
    __shared__ float cs[496 * 8];            // conv outputs [pos][ch]
    const int b = blockIdx.x;
    const int tid = threadIdx.x;
    // 784 floats = 196 x 16B: async direct-to-LDS copy
    if (tid < 196) async_ld_b128(&xs[tid * 4], x + (size_t)b * 784 + tid * 4);
    for (int i = 784 + tid; i < 904; i += 256) xs[i] = 0.0f;
    wait_asynccnt0();
    __syncthreads();

    const int lane = tid & 31;
    const int wave = tid >> 5;
    const int m = lane & 15;   // A row / B col within tile
    const int h = lane >> 4;   // lane half

    // B fragments (weights), 2 K-chunks; w layout [8][49]; clamped + select
    v16h bf[2];
#pragma unroll
    for (int c = 0; c < 2; ++c) {
#pragma unroll
        for (int i = 0; i < 16; ++i) {
            int k = c * 32 + h * 16 + i;
            float v = w[imin(m, 7) * 49 + imin(k, 48)];
            bf[c][i] = (_Float16)((m < 8 && k < 49) ? v : 0.0f);
        }
    }

    // hoisted k -> ky*28+kx offsets (depend only on lane half, not tile)
    int koff[2][16];
#pragma unroll
    for (int c = 0; c < 2; ++c)
#pragma unroll
        for (int i = 0; i < 16; ++i) {
            int k = c * 32 + h * 16 + i;
            koff[c][i] = (k / 7) * 28 + (k % 7);
        }

    for (int t = wave; t < 31; t += 8) {   // wave-uniform tile loop
        int r = t * 16 + m;                 // conv position, < 496
        int base = (r / 22) * 28 + (r % 22);
        v8f acc = {};
#pragma unroll
        for (int c = 0; c < 2; ++c) {
            v16h af;
#pragma unroll
            for (int i = 0; i < 16; ++i)
                af[i] = (_Float16)xs[base + koff[c][i]];
            acc = __builtin_amdgcn_wmma_f32_16x16x32_f16(false, af, false, bf[c],
                                                         (short)0, acc, false, false);
        }
        // D layout: element rr -> row rr+8*h, col m ; rows always < 496
        if (m < 8) {
#pragma unroll
            for (int rr = 0; rr < 8; ++rr)
                cs[(t * 16 + rr + 8 * h) * 8 + m] = acc[rr];
        }
    }
    __syncthreads();

    // 2x2 maxpool (22x22 -> 11x11) + bias + relu; h1 layout [B][8][11][11]
    for (int o = tid; o < 968; o += 256) {
        int ch = o / 121;
        int rem = o % 121;
        int i = rem / 11, j = rem % 11;
        int r0 = (2 * i) * 22 + 2 * j;
        float v0 = cs[r0 * 8 + ch], v1 = cs[(r0 + 1) * 8 + ch];
        float v2 = cs[(r0 + 22) * 8 + ch], v3 = cs[(r0 + 23) * 8 + ch];
        float v = fmaxf(fmaxf(v0, v1), fmaxf(v2, v3)) + bias[ch];
        h1[(size_t)b * 968 + o] = fmaxf(v, 0.0f);
    }
}

// ---------------------------------------------------------------------------
// K2: convk (5x5, 8->10) + pool + relu, fp32 VALU (N=10 -> WMMA padding waste)
// ---------------------------------------------------------------------------
__global__ __launch_bounds__(128) void k2_convk(const float* __restrict__ h1,
                                                const float* __restrict__ w,
                                                const float* __restrict__ bias,
                                                float* __restrict__ h2) {
    __shared__ float hs[968];
    __shared__ float ws[2000];
    __shared__ float bs[10];
    const int b = blockIdx.x, tid = threadIdx.x;
    for (int i = tid; i < 968; i += 128) hs[i] = h1[(size_t)b * 968 + i];
    for (int i = tid; i < 2000; i += 128) ws[i] = w[i];
    if (tid < 10) bs[tid] = bias[tid];
    __syncthreads();
    if (tid < 90) {   // output [10][3][3]
        int ch = tid / 9, rem = tid % 9, i = rem / 3, j = rem % 3;
        float mx = -INFINITY;
        for (int di = 0; di < 2; ++di)
            for (int dj = 0; dj < 2; ++dj) {
                int oy = 2 * i + di, ox = 2 * j + dj;   // conv out 7x7
                float s = 0.0f;
                for (int ci = 0; ci < 8; ++ci)
#pragma unroll
                    for (int ky = 0; ky < 5; ++ky)
#pragma unroll
                        for (int kx = 0; kx < 5; ++kx)
                            s += hs[ci * 121 + (oy + ky) * 11 + (ox + kx)] *
                                 ws[((ch * 8 + ci) * 5 + ky) * 5 + kx];
                mx = fmaxf(mx, s);
            }
        h2[(size_t)b * 90 + tid] = fmaxf(mx + bs[ch], 0.0f);
    }
}

// ---------------------------------------------------------------------------
// K3: kf1 (90->32) + relu + kf2 (32->4) fused, WMMA. One wave per 16 samples.
//     All guarded loads use clamped addresses + branchless select.
// ---------------------------------------------------------------------------
__global__ __launch_bounds__(32) void k3_kf(const float* __restrict__ h2,
                                            const float* __restrict__ w1,
                                            const float* __restrict__ b1,
                                            const float* __restrict__ w2,
                                            const float* __restrict__ b2,
                                            float* __restrict__ kern) {
    __shared__ _Float16 hsf[16 * 32];   // relu(kf1) result, A-staging for kf2
    const int blk = blockIdx.x;
    const int lane = threadIdx.x & 31;
    const int m = lane & 15, h = lane >> 4;
    const int s = blk * 16 + m;

    v8f acc0 = {}, acc1 = {};
#pragma unroll
    for (int c = 0; c < 3; ++c) {        // K = 90 pad 96
        v16h af, bf0, bf1;
#pragma unroll
        for (int i = 0; i < 16; ++i) {
            int ka = c * 32 + h * 16 + i;
            float va = h2[(size_t)s * 90 + imin(ka, 89)];
            af[i] = (_Float16)((ka < 90) ? va : 0.0f);
            int kb = c * 32 + h * 16 + i;
            int kbc = imin(kb, 89);
            float v0 = w1[m * 90 + kbc];
            float v1 = w1[(16 + m) * 90 + kbc];
            bf0[i] = (_Float16)((kb < 90) ? v0 : 0.0f);
            bf1[i] = (_Float16)((kb < 90) ? v1 : 0.0f);
        }
        acc0 = __builtin_amdgcn_wmma_f32_16x16x32_f16(false, af, false, bf0,
                                                      (short)0, acc0, false, false);
        acc1 = __builtin_amdgcn_wmma_f32_16x16x32_f16(false, af, false, bf1,
                                                      (short)0, acc1, false, false);
    }
#pragma unroll
    for (int rr = 0; rr < 8; ++rr) {
        int row = rr + 8 * h;
        hsf[row * 32 + m]      = (_Float16)fmaxf(acc0[rr] + b1[m], 0.0f);
        hsf[row * 32 + 16 + m] = (_Float16)fmaxf(acc1[rr] + b1[16 + m], 0.0f);
    }
    __syncthreads();

    // kf2: [16,32] x [32, 4 pad 16]
    v16h af2, bf2;
#pragma unroll
    for (int i = 0; i < 16; ++i) {
        af2[i] = hsf[m * 32 + h * 16 + i];
        int k = h * 16 + i;                       // < 32 always
        float v = w2[imin(m, 3) * 32 + k];
        bf2[i] = (_Float16)((m < 4) ? v : 0.0f);
    }
    v8f acc2 = {};
    acc2 = __builtin_amdgcn_wmma_f32_16x16x32_f16(false, af2, false, bf2,
                                                  (short)0, acc2, false, false);
    if (m < 4) {
        float bb = b2[m];
#pragma unroll
        for (int rr = 0; rr < 8; ++rr) {
            int row = rr + 8 * h;
            kern[(size_t)(blk * 16 + row) * 4 + m] = acc2[rr] + bb;
        }
    }
}

// ---------------------------------------------------------------------------
// K4: per-image 2x2 conv (27x27) + pool (13x13) + relu; async LDS staging
// ---------------------------------------------------------------------------
__global__ __launch_bounds__(256) void k4_imgconv(const float* __restrict__ x,
                                                  const float* __restrict__ kern,
                                                  float* __restrict__ y1) {
    __shared__ __align__(16) float xs[784];
    const int b = blockIdx.x, tid = threadIdx.x;
    if (tid < 196) async_ld_b128(&xs[tid * 4], x + (size_t)b * 784 + tid * 4);
    float k0 = kern[b * 4 + 0], k1 = kern[b * 4 + 1];
    float k2 = kern[b * 4 + 2], k3 = kern[b * 4 + 3];
    wait_asynccnt0();
    __syncthreads();
    if (tid < 169) {
        int i = tid / 13, j = tid % 13;
        float mx = -INFINITY;
#pragma unroll
        for (int di = 0; di < 2; ++di)
#pragma unroll
            for (int dj = 0; dj < 2; ++dj) {
                int oy = 2 * i + di, ox = 2 * j + dj;
                float v = xs[oy * 28 + ox] * k0 + xs[oy * 28 + ox + 1] * k1 +
                          xs[(oy + 1) * 28 + ox] * k2 + xs[(oy + 1) * 28 + ox + 1] * k3;
                mx = fmaxf(mx, v);
            }
        y1[(size_t)b * 169 + tid] = fmaxf(mx, 0.0f);
    }
}

// ---------------------------------------------------------------------------
// K5: conv2 (5x5, 1->20) + pool + relu, fp32 VALU
// ---------------------------------------------------------------------------
__global__ __launch_bounds__(128) void k5_conv2(const float* __restrict__ y1,
                                                const float* __restrict__ w,
                                                const float* __restrict__ bias,
                                                float* __restrict__ y2) {
    __shared__ float ys[169];
    __shared__ float ws[500];
    __shared__ float bs[20];
    const int b = blockIdx.x, tid = threadIdx.x;
    for (int i = tid; i < 169; i += 128) ys[i] = y1[(size_t)b * 169 + i];
    for (int i = tid; i < 500; i += 128) ws[i] = w[i];
    if (tid < 20) bs[tid] = bias[tid];
    __syncthreads();
    for (int o = tid; o < 320; o += 128) {   // [20][4][4]
        int ch = o >> 4, rem = o & 15, i = rem >> 2, j = rem & 3;
        float mx = -INFINITY;
        for (int di = 0; di < 2; ++di)
            for (int dj = 0; dj < 2; ++dj) {
                int oy = 2 * i + di, ox = 2 * j + dj;   // conv out 9x9
                float s = 0.0f;
#pragma unroll
                for (int ky = 0; ky < 5; ++ky)
#pragma unroll
                    for (int kx = 0; kx < 5; ++kx)
                        s += ys[(oy + ky) * 13 + ox + kx] * ws[ch * 25 + ky * 5 + kx];
                mx = fmaxf(mx, s);
            }
        y2[(size_t)b * 320 + o] = fmaxf(mx + bs[ch], 0.0f);
    }
}

// ---------------------------------------------------------------------------
// K6: fc1 (320->50) + relu + fc2 (50->10) + log_softmax fused, WMMA.
//     One wave per 16 samples. fc1: K=320 (10 chunks), N=50 pad 64 (4 tiles).
//     A and fc1-B fragments load as float4 (contiguous 8/16-float runs).
// ---------------------------------------------------------------------------
__global__ __launch_bounds__(32) void k6_fc(const float* __restrict__ y2,
                                            const float* __restrict__ w1,
                                            const float* __restrict__ b1,
                                            const float* __restrict__ w2,
                                            const float* __restrict__ b2,
                                            float* __restrict__ out) {
    __shared__ _Float16 a1[16 * 64];    // relu(fc1), zero-padded to K=64
    __shared__ float logits[16 * 16];
    const int blk = blockIdx.x;
    const int lane = threadIdx.x & 31;
    const int m = lane & 15, h = lane >> 4;
    const int s = blk * 16 + m;

    __builtin_prefetch(w1, 0, 3);   // global_prefetch_b8: weights into cache

    const float4* yrow = (const float4*)(y2 + (size_t)s * 320);

    v8f acc[4] = {};
    for (int c = 0; c < 10; ++c) {
        // A: k = c*32 + h*16 + i, 16 consecutive floats -> 4x float4
        float4 p0 = yrow[c * 8 + h * 4];
        float4 p1 = yrow[c * 8 + h * 4 + 1];
        float4 p2 = yrow[c * 8 + h * 4 + 2];
        float4 p3 = yrow[c * 8 + h * 4 + 3];
        v16h af;
        af[0] = (_Float16)p0.x;  af[1] = (_Float16)p0.y;
        af[2] = (_Float16)p0.z;  af[3] = (_Float16)p0.w;
        af[4] = (_Float16)p1.x;  af[5] = (_Float16)p1.y;
        af[6] = (_Float16)p1.z;  af[7] = (_Float16)p1.w;
        af[8] = (_Float16)p2.x;  af[9] = (_Float16)p2.y;
        af[10] = (_Float16)p2.z; af[11] = (_Float16)p2.w;
        af[12] = (_Float16)p3.x; af[13] = (_Float16)p3.y;
        af[14] = (_Float16)p3.z; af[15] = (_Float16)p3.w;
#pragma unroll
        for (int nt = 0; nt < 4; ++nt) {
            int n = nt * 16 + m;
            bool nv = (n < 50);
            // B: k = c*32 + h*16 + i, 16 consecutive floats -> 4x float4
            const float4* wrow =
                (const float4*)(w1 + (size_t)imin(n, 49) * 320 + c * 32 + h * 16);
            float4 q0 = wrow[0], q1 = wrow[1], q2 = wrow[2], q3 = wrow[3];
            v16h bf;
            bf[0] = (_Float16)(nv ? q0.x : 0.0f);  bf[1] = (_Float16)(nv ? q0.y : 0.0f);
            bf[2] = (_Float16)(nv ? q0.z : 0.0f);  bf[3] = (_Float16)(nv ? q0.w : 0.0f);
            bf[4] = (_Float16)(nv ? q1.x : 0.0f);  bf[5] = (_Float16)(nv ? q1.y : 0.0f);
            bf[6] = (_Float16)(nv ? q1.z : 0.0f);  bf[7] = (_Float16)(nv ? q1.w : 0.0f);
            bf[8] = (_Float16)(nv ? q2.x : 0.0f);  bf[9] = (_Float16)(nv ? q2.y : 0.0f);
            bf[10] = (_Float16)(nv ? q2.z : 0.0f); bf[11] = (_Float16)(nv ? q2.w : 0.0f);
            bf[12] = (_Float16)(nv ? q3.x : 0.0f); bf[13] = (_Float16)(nv ? q3.y : 0.0f);
            bf[14] = (_Float16)(nv ? q3.z : 0.0f); bf[15] = (_Float16)(nv ? q3.w : 0.0f);
            acc[nt] = __builtin_amdgcn_wmma_f32_16x16x32_f16(false, af, false, bf,
                                                             (short)0, acc[nt], false, false);
        }
    }
#pragma unroll
    for (int nt = 0; nt < 4; ++nt) {
        int n = nt * 16 + m;
        float bb = b1[imin(n, 49)];
#pragma unroll
        for (int rr = 0; rr < 8; ++rr) {
            int row = rr + 8 * h;
            float v = (n < 50) ? fmaxf(acc[nt][rr] + bb, 0.0f) : 0.0f;
            a1[row * 64 + n] = (_Float16)v;
        }
    }
    __syncthreads();

    // fc2: K=50 pad 64 (2 chunks), N=10 pad 16; clamped + select B loads
    v8f acc2 = {};
#pragma unroll
    for (int c = 0; c < 2; ++c) {
        v16h af, bf;
#pragma unroll
        for (int i = 0; i < 16; ++i) {
            af[i] = a1[m * 64 + c * 32 + h * 16 + i];
            int k = c * 32 + h * 16 + i;
            float v = w2[imin(m, 9) * 50 + imin(k, 49)];
            bf[i] = (_Float16)((m < 10 && k < 50) ? v : 0.0f);
        }
        acc2 = __builtin_amdgcn_wmma_f32_16x16x32_f16(false, af, false, bf,
                                                      (short)0, acc2, false, false);
    }
    {
        float bb = b2[imin(m, 9)];
#pragma unroll
        for (int rr = 0; rr < 8; ++rr) {
            int row = rr + 8 * h;
            logits[row * 16 + m] = acc2[rr] + ((m < 10) ? bb : 0.0f);
        }
    }
    __syncthreads();

    if (lane < 16) {
        float mx = -INFINITY;
        for (int n = 0; n < 10; ++n) mx = fmaxf(mx, logits[lane * 16 + n]);
        float sum = 0.0f;
        for (int n = 0; n < 10; ++n) sum += expf(logits[lane * 16 + n] - mx);
        float lse = logf(sum);
        for (int n = 0; n < 10; ++n)
            out[(size_t)(blk * 16 + lane) * 10 + n] = logits[lane * 16 + n] - mx - lse;
    }
}

// ---------------------------------------------------------------------------
extern "C" void kernel_launch(void* const* d_in, const int* in_sizes, int n_in,
                              void* d_out, int out_size, void* d_ws, size_t ws_size,
                              hipStream_t stream) {
    const float* x       = (const float*)d_in[0];
    const float* conv1_w = (const float*)d_in[1];
    const float* conv1_b = (const float*)d_in[2];
    const float* convk_w = (const float*)d_in[3];
    const float* convk_b = (const float*)d_in[4];
    const float* kf1_w   = (const float*)d_in[5];
    const float* kf1_b   = (const float*)d_in[6];
    const float* kf2_w   = (const float*)d_in[7];
    const float* kf2_b   = (const float*)d_in[8];
    const float* conv2_w = (const float*)d_in[9];
    const float* conv2_b = (const float*)d_in[10];
    const float* fc1_w   = (const float*)d_in[11];
    const float* fc1_b   = (const float*)d_in[12];
    const float* fc2_w   = (const float*)d_in[13];
    const float* fc2_b   = (const float*)d_in[14];
    float* out = (float*)d_out;

    const int B = in_sizes[0] / 784;   // 8192

    float* ws   = (float*)d_ws;
    float* h1   = ws;                        // [B][8][11][11] = B*968
    float* h2   = h1 + (size_t)B * 968;      // [B][90]
    float* kern = h2 + (size_t)B * 90;       // [B][4]
    float* y1   = kern + (size_t)B * 4;      // [B][169]
    float* yy2  = y1 + (size_t)B * 169;      // [B][320]

    k1_conv1<<<B, 256, 0, stream>>>(x, conv1_w, conv1_b, h1);
    k2_convk<<<B, 128, 0, stream>>>(h1, convk_w, convk_b, h2);
    k3_kf<<<B / 16, 32, 0, stream>>>(h2, kf1_w, kf1_b, kf2_w, kf2_b, kern);
    k4_imgconv<<<B, 256, 0, stream>>>(x, kern, y1);
    k5_conv2<<<B, 128, 0, stream>>>(y1, conv2_w, conv2_b, yy2);
    k6_fc<<<B / 16, 32, 0, stream>>>(yy2, fc1_w, fc1_b, fc2_w, fc2_b, out);
}
